// SketchVAE_v2_9835475108283
// MI455X (gfx1250) — matrix operand order repaired and checked by reading, not compiled, not run on hardware
//
#include <hip/hip_runtime.h>
#include <hip/hip_bf16.h>
#include <stdint.h>

// ---------------------------------------------------------------------------
// SketchVAE forward for MI455X (gfx1250, wave32, WMMA f16->f32 16x16x32)
// ---------------------------------------------------------------------------

typedef __attribute__((ext_vector_type(16))) _Float16 v16h;
typedef __attribute__((ext_vector_type(8)))  float    v8f;

#define BDIM 128
#define SDIM 256
#define HDIM 512
#define VD   130
#define PV   10

// ---------------- WMMA core helpers ----------------------------------------

__device__ inline v8f wmma32(v16h a, v16h b, v8f c) {
  // D = A(16x32 f16) x B(32x16 f16) + C(16x16 f32)
  return __builtin_amdgcn_wmma_f32_16x16x32_f16(false, a, false, b, (short)0, c, false, false);
}

// A-matrix (16x32 f16), row-major source with leading dim `ld` (elements).
// Layout per ISA: lanes 0-15 -> rows, K {0..7,16..23}; lanes 16-31 -> K {8..15,24..31}.
__device__ inline v16h load_A_f16(const _Float16* A, size_t ld, int k0, int lane) {
  int row = lane & 15;
  int off = (lane >> 4) ? 8 : 0;
  const _Float16* p = A + (size_t)row * ld + k0 + off;
  v16h a;
#pragma unroll
  for (int j = 0; j < 8; j++) { a[j] = p[j]; a[8 + j] = p[16 + j]; }
  return a;
}

__device__ inline v16h load_A_f32cvt(const float* A, size_t ld, int k0, int lane) {
  int row = lane & 15;
  int off = (lane >> 4) ? 8 : 0;
  const float* p = A + (size_t)row * ld + k0 + off;
  v16h a;
#pragma unroll
  for (int j = 0; j < 8; j++) { a[j] = (_Float16)p[j]; a[8 + j] = (_Float16)p[16 + j]; }
  return a;
}

// B-matrix (32x16 f16): element (k,n) = W[n][k] with W row-major [N][K].
// lanes 0-15: N=lane, K {0..7,16..23} ; lanes 16-31: N=lane-16, K {8..15,24..31}.
__device__ inline v16h load_B_f16(const _Float16* W, int ldk, int n0, int k0, int lane) {
  int col  = lane & 15;
  int koff = k0 + ((lane >> 4) ? 8 : 0);
  const _Float16* p = W + (size_t)(n0 + col) * ldk + koff;
  v16h b;
#pragma unroll
  for (int j = 0; j < 8; j++) { b[j] = p[j]; b[8 + j] = p[16 + j]; }
  return b;
}

__device__ inline v8f mm_f32A(const float* A, size_t lda, const _Float16* W, int ldw,
                              int n0, int K, int lane, v8f acc) {
  for (int k = 0; k < K; k += 32)
    acc = wmma32(load_A_f32cvt(A, lda, k, lane), load_B_f16(W, ldw, n0, k, lane), acc);
  return acc;
}

__device__ inline void zero23(v8f a[2][3]) {
  v8f z = {};
#pragma unroll
  for (int i = 0; i < 2; i++)
#pragma unroll
    for (int j = 0; j < 3; j++) a[i][j] = z;
}

// Fused GRU-gate GEMM: one shared A fragment feeds r/z/n gate tiles for BOTH
// o-tiles of this wave -> 6 WMMAs per A-load (6x less A traffic, deeper
// independent-WMMA window for the XDL pipe).
__device__ inline void mm_gru6(const _Float16* A, size_t lda, const _Float16* W, int ldw,
                               int o0a, int o0b, int K, int lane, v8f acc[2][3]) {
  for (int k = 0; k < K; k += 32) {
    v16h a = load_A_f16(A, lda, k, lane);
#pragma unroll
    for (int g = 0; g < 3; g++) {
      acc[0][g] = wmma32(a, load_B_f16(W, ldw, g * 512 + o0a, k, lane), acc[0][g]);
      acc[1][g] = wmma32(a, load_B_f16(W, ldw, g * 512 + o0b, k, lane), acc[1][g]);
    }
  }
}

__device__ inline float sigmf(float x) { return 1.f / (1.f + __expf(-x)); }

// GRU gate combine for one 16x16 output tile (o = absolute hidden index of this col).
// acc element e lives at (m = e + 8*(lane>=16), o). Updates f16 hidden in LDS;
// optional f16 global out (per-row stride outGMS) and f32 out (outFMS).
__device__ inline void gru_combine_tile(
    v8f ir, v8f iz, v8f inn, v8f hr, v8f hz, v8f hn,
    const float* bih, const float* bhh, int o, int lane,
    _Float16* hS, const int* lenS, int t,
    _Float16* outG, size_t outGMS, float* outF, size_t outFMS) {
  int half = lane >> 4;
  float br = bih[o], bz = bih[512 + o], bn = bih[1024 + o];
  float cr = bhh[o], cz = bhh[512 + o], cn = bhh[1024 + o];
#pragma unroll
  for (int e = 0; e < 8; e++) {
    int m = e + half * 8;
    float hold = (float)hS[m * 512 + o];
    float r = sigmf(ir[e] + br + hr[e] + cr);
    float z = sigmf(iz[e] + bz + hz[e] + cz);
    float n = tanhf(inn[e] + bn + r * (hn[e] + cn));
    float hv = (1.f - z) * n + z * hold;
    if (lenS && !(t < lenS[m])) hv = hold;           // masked variable-length update
    hS[m * 512 + o] = (_Float16)hv;
    if (outG) outG[(size_t)m * outGMS + o] = (_Float16)hv;
    if (outF) outF[(size_t)m * outFMS + o] = hv;
  }
}

// ---------------- weight f32 -> f16 packing (with row padding) --------------

__global__ void __launch_bounds__(256) conv_kernel(const float* src, _Float16* dst,
                                                   int rows, int srcld, int coloff,
                                                   int cols, int totRows) {
  size_t i = (size_t)blockIdx.x * 256 + threadIdx.x;
  size_t n = (size_t)totRows * cols;
  if (i >= n) return;
  int r = (int)(i / cols), c = (int)(i % cols);
  float v = (r < rows) ? src[(size_t)r * srcld + coloff + c] : 0.f;
  dst[i] = (_Float16)v;
}

// ---------------- encoder layer 0 (4 scans: p-fwd, p-bwd, r-fwd, r-bwd) -----

struct EncL0Args {
  const int* px; const float* rx; const float* pemb; const int* len;
  const float*    wih[4];
  const _Float16* whh[4];
  const float*    bih[4];
  const float*    bhh[4];
  _Float16*       out[4];   // enc buffer + dir*512 (layer-1 input, f16 [B,S,1024])
  float*          hT[4];    // n_x base + slot*512 (row stride 2048)
  int kx[4]; int dir[4]; int masked[4]; int isPitch[4];
};

__global__ void __launch_bounds__(512) enc_l0_kernel(EncL0Args A) {
  const int S = SDIM;
  int scan = blockIdx.x >> 3, mb = blockIdx.x & 7;
  int tid = threadIdx.x, wv = tid >> 5, lane = tid & 31;
  __shared__ _Float16 hS[16 * 512];
  __shared__ float xS[16 * 16];
  __shared__ int lenS[16];
  for (int i = tid; i < 16 * 512; i += 512) hS[i] = (_Float16)0.f;
  if (tid < 16) lenS[tid] = A.masked[scan] ? A.len[mb * 16 + tid] : 0x7fffffff;
  __syncthreads();
  const float* wih = A.wih[scan];
  const _Float16* whh = A.whh[scan];
  const float* bih = A.bih[scan];
  const float* bhh = A.bhh[scan];
  _Float16* outB = A.out[scan];
  float* hT = A.hT[scan];
  int kx = A.kx[scan], dir = A.dir[scan], isP = A.isPitch[scan];
  int o0a = wv * 16, o0b = (wv + 16) * 16;

  for (int sstep = 0; sstep < S; sstep++) {
    int t = dir ? (S - 1 - sstep) : sstep;
    if (tid < 16) {                       // stage x_t [16][kx]
      int b = mb * 16 + tid;
      if (isP) {
        int ix = A.px[(size_t)b * S + t];
#pragma unroll
        for (int k = 0; k < PV; k++) xS[tid * 16 + k] = A.pemb[ix * PV + k];
      } else {
        for (int k = 0; k < 3; k++) xS[tid * 16 + k] = A.rx[((size_t)b * S + t) * 3 + k];
      }
    }
    __syncthreads();
    v8f aH[2][3];
    zero23(aH);
    mm_gru6(hS, 512, whh, 512, o0a, o0b, 512, lane, aH);   // h @ Whh^T (WMMA)
    float aI[2][3][8];
#pragma unroll
    for (int ti = 0; ti < 2; ti++) {
      int o = (ti ? o0b : o0a) + (lane & 15);
      int half = lane >> 4;
#pragma unroll
      for (int g = 0; g < 3; g++)
#pragma unroll
        for (int e = 0; e < 8; e++) aI[ti][g][e] = 0.f;
      for (int k = 0; k < kx; k++) {      // tiny-K input projection (VALU)
        float w0 = wih[(size_t)o * kx + k];
        float w1 = wih[(size_t)(512 + o) * kx + k];
        float w2 = wih[(size_t)(1024 + o) * kx + k];
#pragma unroll
        for (int e = 0; e < 8; e++) {
          float xv = xS[(e + half * 8) * 16 + k];
          aI[ti][0][e] += xv * w0;
          aI[ti][1][e] += xv * w1;
          aI[ti][2][e] += xv * w2;
        }
      }
    }
    __syncthreads();
    _Float16* outRow = outB + ((size_t)(mb * 16) * S + t) * 1024;
#pragma unroll
    for (int ti = 0; ti < 2; ti++) {
      int o = (ti ? o0b : o0a) + (lane & 15);
      v8f vr, vz, vn;
#pragma unroll
      for (int e = 0; e < 8; e++) { vr[e] = aI[ti][0][e]; vz[e] = aI[ti][1][e]; vn[e] = aI[ti][2][e]; }
      gru_combine_tile(vr, vz, vn, aH[ti][0], aH[ti][1], aH[ti][2],
                       bih, bhh, o, lane, hS, lenS, t,
                       outRow, (size_t)S * 1024, nullptr, 0);
    }
    __syncthreads();
  }
  for (int i = tid; i < 16 * 512; i += 512) {
    int m = i >> 9, o = i & 511;
    hT[(size_t)(mb * 16 + m) * 2048 + o] = (float)hS[i];
  }
}

// ---------------- encoder layer 1 (input K=1024 via WMMA) -------------------

struct EncL1Args {
  const _Float16* xin[4];
  const _Float16* wih[4];
  const _Float16* whh[4];
  const float* bih[4]; const float* bhh[4];
  const int* len; int masked[4]; int dir[4];
  float* hT[4];
};

__global__ void __launch_bounds__(512) enc_l1_kernel(EncL1Args A) {
  const int S = SDIM;
  int scan = blockIdx.x >> 3, mb = blockIdx.x & 7;
  int tid = threadIdx.x, wv = tid >> 5, lane = tid & 31;
  __shared__ _Float16 hS[16 * 512];
  __shared__ int lenS[16];
  for (int i = tid; i < 16 * 512; i += 512) hS[i] = (_Float16)0.f;
  if (tid < 16) lenS[tid] = A.masked[scan] ? A.len[mb * 16 + tid] : 0x7fffffff;
  __syncthreads();
  const _Float16* X = A.xin[scan];
  const _Float16* wih = A.wih[scan];
  const _Float16* whh = A.whh[scan];
  const float* bih = A.bih[scan];
  const float* bhh = A.bhh[scan];
  float* hT = A.hT[scan];
  int dir = A.dir[scan];
  size_t ldx = (size_t)S * 1024;
  int o0a = wv * 16, o0b = (wv + 16) * 16;
  for (int sstep = 0; sstep < S; sstep++) {
    int t = dir ? (S - 1 - sstep) : sstep;
    const _Float16* Xrow = X + ((size_t)(mb * 16) * S + t) * 1024;
    v8f aI[2][3], aH[2][3];
    zero23(aI); zero23(aH);
    mm_gru6(Xrow, ldx, wih, 1024, o0a, o0b, 1024, lane, aI);  // x @ Wih^T
    mm_gru6(hS, 512, whh, 512, o0a, o0b, 512, lane, aH);      // h @ Whh^T
    __syncthreads();
#pragma unroll
    for (int ti = 0; ti < 2; ti++) {
      int o = (ti ? o0b : o0a) + (lane & 15);
      gru_combine_tile(aI[ti][0], aI[ti][1], aI[ti][2], aH[ti][0], aH[ti][1], aH[ti][2],
                       bih, bhh, o, lane, hS, lenS, t, nullptr, 0, nullptr, 0);
    }
    __syncthreads();
  }
  for (int i = tid; i < 16 * 512; i += 512) {
    int m = i >> 9, o = i & 511;
    hT[(size_t)(mb * 16 + m) * 2048 + o] = (float)hS[i];
  }
}

// ---------------- generic linear: C = act(A[M,K]f32 x W[N,K]f16 + b) --------

__global__ void __launch_bounds__(256) lin_kernel(const float* A, const _Float16* W,
                                                  const float* bias, float* C,
                                                  int M, int N, int K, int act) {
  int wv = threadIdx.x >> 5, lane = threadIdx.x & 31;
  int tile = blockIdx.x * 8 + wv;
  int Nt = N >> 4;
  int total = (M >> 4) * Nt;
  if (tile >= total) return;                  // wave-uniform exit
  int tm = tile / Nt, tn = tile % Nt;
  v8f acc = {};
  acc = mm_f32A(A + (size_t)tm * 16 * K, (size_t)K, W, K, tn * 16, K, lane, acc);
  int col = lane & 15, half = lane >> 4, n = tn * 16 + col;
  float bb = bias[n];
#pragma unroll
  for (int e = 0; e < 8; e++) {
    int m = tm * 16 + e + half * 8;
    float v = acc[e] + bb;
    if (act == 1) {                           // SELU
      const float sc = 1.0507009873554805f, al = 1.6732632423543772f;
      v = v > 0.f ? sc * v : sc * al * (__expf(v) - 1.f);
    } else if (act == 2) {                    // exp
      v = __expf(v);
    }
    C[(size_t)m * N + n] = v;
  }
}

// ---------------- reparameterized sample ------------------------------------

__global__ void __launch_bounds__(256) sample_kernel(const float* pmu, const float* psc,
                                                     const float* rmu, const float* rsc,
                                                     const float* np_, const float* nr_,
                                                     float* z) {
  int i = blockIdx.x * 256 + threadIdx.x;
  if (i < 128 * 128) {
    int b = i >> 7, k = i & 127;
    z[b * 256 + k]       = pmu[i] + psc[i] * np_[i];
    z[b * 256 + 128 + k] = rmu[i] + rsc[i] * nr_[i];
  }
}

// ---------------- beat decoder: 2-layer GRU, 16 steps -----------------------

__global__ void __launch_bounds__(512) beat_kernel(
    const float* hb, const float* beat0,
    const float* wih0, const _Float16* whh0, const float* bih0, const float* bhh0,
    const _Float16* wih1, const _Float16* whh1, const float* bih1, const float* bhh1,
    float* beatOut) {
  int mb = blockIdx.x, tid = threadIdx.x, wv = tid >> 5, lane = tid & 31;
  __shared__ _Float16 h0S[16 * 512], h1S[16 * 512];
  for (int i = tid; i < 16 * 512; i += 512) {
    int m = i >> 9, o = i & 511;
    size_t row = (size_t)(mb * 16 + m);
    h0S[i] = (_Float16)hb[row * 1024 + o];
    h1S[i] = (_Float16)hb[row * 1024 + 512 + o];
  }
  __syncthreads();
  float x0 = beat0[0];
  int o0a = wv * 16, o0b = (wv + 16) * 16;
  for (int t = 0; t < 16; t++) {
    v8f aH[2][3];
    zero23(aH);
    mm_gru6(h0S, 512, whh0, 512, o0a, o0b, 512, lane, aH);
    __syncthreads();
#pragma unroll
    for (int ti = 0; ti < 2; ti++) {
      int o = (ti ? o0b : o0a) + (lane & 15);
      float w0 = x0 * wih0[o], w1 = x0 * wih0[512 + o], w2 = x0 * wih0[1024 + o];
      v8f vr, vz, vn;
#pragma unroll
      for (int e = 0; e < 8; e++) { vr[e] = w0; vz[e] = w1; vn[e] = w2; }
      gru_combine_tile(vr, vz, vn, aH[ti][0], aH[ti][1], aH[ti][2],
                       bih0, bhh0, o, lane, h0S, nullptr, 0, nullptr, 0, nullptr, 0);
    }
    __syncthreads();
    v8f bI[2][3], bH[2][3];
    zero23(bI); zero23(bH);
    mm_gru6(h0S, 512, wih1, 512, o0a, o0b, 512, lane, bI);
    mm_gru6(h1S, 512, whh1, 512, o0a, o0b, 512, lane, bH);
    __syncthreads();
    float* outRow = beatOut + ((size_t)(mb * 16) * 16 + t) * 512;
#pragma unroll
    for (int ti = 0; ti < 2; ti++) {
      int o = (ti ? o0b : o0a) + (lane & 15);
      gru_combine_tile(bI[ti][0], bI[ti][1], bI[ti][2], bH[ti][0], bH[ti][1], bH[ti][2],
                       bih1, bhh1, o, lane, h1S, nullptr, 0, nullptr, 0,
                       outRow, (size_t)16 * 512);
    }
    __syncthreads();
  }
}

// ---------------- tick decoder: 16 beats x 16 ticks, argmax feedback --------

__global__ void __launch_bounds__(512) tick_kernel(
    const float* hTick, const float* cTick, const float* tick0, const float* demb,
    const float* wih0, const _Float16* wihCt, const _Float16* whh0,
    const float* bih0, const float* bhh0,
    const _Float16* wih1, const _Float16* whh1, const float* bih1, const float* bhh1,
    const _Float16* w2n, const float* b2n, float* recon) {
  int mb = blockIdx.x, tid = threadIdx.x, wv = tid >> 5, lane = tid & 31;
  __shared__ _Float16 h0S[8192], h1S[8192], ctS[8192];
  __shared__ float lgS[16 * 144];
  __shared__ float yS[16 * PV];
  if (tid < 16 * PV) yS[tid] = tick0[tid % PV];
  __syncthreads();
  int o0a = wv * 16, o0b = (wv + 16) * 16;
  for (int beat = 0; beat < 16; beat++) {
    for (int i = tid; i < 8192; i += 512) {
      int m = i >> 9, o = i & 511;
      size_t row = (size_t)(mb * 16 + m) * 16 + beat;
      h0S[i] = (_Float16)hTick[row * 1024 + o];
      h1S[i] = (_Float16)hTick[row * 1024 + 512 + o];
      ctS[i] = (_Float16)cTick[row * 512 + o];
    }
    __syncthreads();
    // c_tick input projection: constant for 16 ticks -> hoist, hold in VGPRs
    v8f aCt[2][3];
    zero23(aCt);
    mm_gru6(ctS, 512, wihCt, 512, o0a, o0b, 512, lane, aCt);
    for (int tick = 0; tick < 16; tick++) {
      v8f aH[2][3];
      zero23(aH);
      mm_gru6(h0S, 512, whh0, 512, o0a, o0b, 512, lane, aH);
      float aY[2][3][8];
#pragma unroll
      for (int ti = 0; ti < 2; ti++) {
        int o = (ti ? o0b : o0a) + (lane & 15);
        int half = lane >> 4;
#pragma unroll
        for (int g = 0; g < 3; g++)
#pragma unroll
          for (int e = 0; e < 8; e++) aY[ti][g][e] = 0.f;
#pragma unroll
        for (int k = 0; k < PV; k++) {       // embedded-token part (x cols 0..9)
          float w0 = wih0[(size_t)o * 522 + k];
          float w1 = wih0[(size_t)(512 + o) * 522 + k];
          float w2 = wih0[(size_t)(1024 + o) * 522 + k];
#pragma unroll
          for (int e = 0; e < 8; e++) {
            float yv = yS[(e + half * 8) * PV + k];
            aY[ti][0][e] += yv * w0; aY[ti][1][e] += yv * w1; aY[ti][2][e] += yv * w2;
          }
        }
      }
      __syncthreads();
#pragma unroll
      for (int ti = 0; ti < 2; ti++) {
        int o = (ti ? o0b : o0a) + (lane & 15);
        v8f vr, vz, vn;
#pragma unroll
        for (int e = 0; e < 8; e++) {
          vr[e] = aCt[ti][0][e] + aY[ti][0][e];
          vz[e] = aCt[ti][1][e] + aY[ti][1][e];
          vn[e] = aCt[ti][2][e] + aY[ti][2][e];
        }
        gru_combine_tile(vr, vz, vn, aH[ti][0], aH[ti][1], aH[ti][2],
                         bih0, bhh0, o, lane, h0S, nullptr, 0, nullptr, 0, nullptr, 0);
      }
      __syncthreads();
      v8f bI[2][3], bH[2][3];
      zero23(bI); zero23(bH);
      mm_gru6(h0S, 512, wih1, 512, o0a, o0b, 512, lane, bI);
      mm_gru6(h1S, 512, whh1, 512, o0a, o0b, 512, lane, bH);
      __syncthreads();
#pragma unroll
      for (int ti = 0; ti < 2; ti++) {
        int o = (ti ? o0b : o0a) + (lane & 15);
        gru_combine_tile(bI[ti][0], bI[ti][1], bI[ti][2], bH[ti][0], bH[ti][1], bH[ti][2],
                         bih1, bhh1, o, lane, h1S, nullptr, 0, nullptr, 0, nullptr, 0);
      }
      __syncthreads();
      // logits = relu(h1 @ W2n^T + b): 9 N-tiles (N padded 130 -> 144),
      // A fragment shared across this wave's tiles.
      {
        int tn0 = wv;
        int tn1 = (wv == 0) ? 8 : -1;
        v8f lg0 = {}, lg1 = {};
        for (int k = 0; k < 512; k += 32) {
          v16h a = load_A_f16(h1S, 512, k, lane);
          lg0 = wmma32(a, load_B_f16(w2n, 512, tn0 * 16, k, lane), lg0);
          if (tn1 >= 0)
            lg1 = wmma32(a, load_B_f16(w2n, 512, tn1 * 16, k, lane), lg1);
        }
#pragma unroll
        for (int rep = 0; rep < 2; rep++) {
          int tn = (rep == 0) ? tn0 : tn1;
          if (tn >= 0) {                     // wave-uniform branch
            v8f lg = (rep == 0) ? lg0 : lg1;
            int col = lane & 15, half = lane >> 4, n = tn * 16 + col;
            float bb = (n < VD) ? b2n[n] : 0.f;
#pragma unroll
            for (int e = 0; e < 8; e++) {
              int m = e + half * 8;
              float v = lg[e] + bb;
              v = v > 0.f ? v : 0.f;
              if (n < VD) {
                lgS[m * 144 + n] = v;
                recon[(size_t)(mb * 16 + m) * (SDIM * VD) + (beat * 16 + tick) * VD + n] = v;
              } else {
                lgS[m * 144 + n] = -1.f;
              }
            }
          }
        }
      }
      __syncthreads();
      if (tid < 16) {                        // argmax (first-max) + emb feedback
        int best = 0; float bv = lgS[tid * 144];
        for (int v = 1; v < VD; v++) {
          float q = lgS[tid * 144 + v];
          if (q > bv) { bv = q; best = v; }
        }
#pragma unroll
        for (int k = 0; k < PV; k++) yS[tid * PV + k] = demb[best * PV + k];
      }
      __syncthreads();
    }
  }
}

// ---------------- host orchestration ----------------------------------------

extern "C" void kernel_launch(void* const* d_in, const int* in_sizes, int n_in,
                              void* d_out, int out_size, void* d_ws, size_t ws_size,
                              hipStream_t stream) {
  (void)in_sizes; (void)n_in; (void)out_size; (void)ws_size;
  const int*   px      = (const int*)d_in[0];
  const float* rx      = (const float*)d_in[1];
  const int*   len     = (const int*)d_in[2];
  const float* noise_p = (const float*)d_in[4];
  const float* noise_r = (const float*)d_in[5];
  const float* pemb    = (const float*)d_in[6];
  const float* demb    = (const float*)d_in[7];
  auto F = [&](int i) { return (const float*)d_in[i]; };

  // workspace carve (~195 MB): f16 weights, f16 layer-1 inputs, f32 activations
  uint8_t* wp = (uint8_t*)d_ws;
  auto carve = [&](size_t bytes) -> void* {
    void* r = (void*)wp;
    wp += (bytes + 255) & ~(size_t)255;
    return r;
  };
  _Float16* whh16[12];
  const int whhSrc[12] = {9, 13, 17, 21, 25, 29, 33, 37, 52, 56, 65, 69};
  for (int i = 0; i < 12; i++) whh16[i] = (_Float16*)carve((size_t)1536 * 512 * 2);
  _Float16* wihL1[4];
  const int wihL1Src[4] = {16, 20, 32, 36};
  for (int i = 0; i < 4; i++) wihL1[i] = (_Float16*)carve((size_t)1536 * 1024 * 2);
  _Float16* beatWih1  = (_Float16*)carve((size_t)1536 * 512 * 2);
  _Float16* tickWihCt = (_Float16*)carve((size_t)1536 * 512 * 2);
  _Float16* tickWih1  = (_Float16*)carve((size_t)1536 * 512 * 2);
  _Float16* t2n16     = (_Float16*)carve((size_t)144 * 512 * 2);
  _Float16* b2th16    = (_Float16*)carve((size_t)1024 * 512 * 2);
  _Float16* b2ti16    = (_Float16*)carve((size_t)512 * 512 * 2);
  _Float16* z2b16     = (_Float16*)carve((size_t)1024 * 256 * 2);
  _Float16* head16[4];
  const int headSrc[4] = {40, 42, 44, 46};
  for (int i = 0; i < 4; i++) head16[i] = (_Float16*)carve((size_t)128 * 2048 * 2);
  _Float16* encP = (_Float16*)carve((size_t)128 * 256 * 1024 * 2);
  _Float16* encR = (_Float16*)carve((size_t)128 * 256 * 1024 * 2);
  float* nPx     = (float*)carve((size_t)128 * 2048 * 4);
  float* nRx     = (float*)carve((size_t)128 * 2048 * 4);
  float* zBuf    = (float*)carve((size_t)128 * 256 * 4);
  float* hbBuf   = (float*)carve((size_t)128 * 1024 * 4);
  float* beatOut = (float*)carve((size_t)128 * 16 * 512 * 4);
  float* hTick   = (float*)carve((size_t)128 * 16 * 1024 * 4);
  float* cTick   = (float*)carve((size_t)128 * 16 * 512 * 4);

  auto conv = [&](const float* src, _Float16* dst, int rows, int srcld, int coloff,
                  int cols, int totRows) {
    size_t n = (size_t)totRows * cols;
    conv_kernel<<<(unsigned)((n + 255) / 256), 256, 0, stream>>>(src, dst, rows, srcld,
                                                                 coloff, cols, totRows);
  };
  for (int i = 0; i < 12; i++) conv(F(whhSrc[i]), whh16[i], 1536, 512, 0, 512, 1536);
  for (int i = 0; i < 4; i++)  conv(F(wihL1Src[i]), wihL1[i], 1536, 1024, 0, 1024, 1536);
  conv(F(55), beatWih1, 1536, 512, 0, 512, 1536);
  conv(F(64), tickWihCt, 1536, 522, 10, 512, 1536);   // x cols 10..521 (c_tick part)
  conv(F(68), tickWih1, 1536, 512, 0, 512, 1536);
  conv(F(72), t2n16, 130, 512, 0, 512, 144);          // pad 130 -> 144 rows
  conv(F(59), b2th16, 1024, 512, 0, 512, 1024);
  conv(F(61), b2ti16, 512, 512, 0, 512, 512);
  conv(F(48), z2b16, 1024, 256, 0, 256, 1024);
  for (int i = 0; i < 4; i++) conv(F(headSrc[i]), head16[i], 128, 2048, 0, 2048, 128);

  // encoder layer 0: scans {p-fwd, p-bwd, r-fwd, r-bwd}, 8 batch-blocks each
  EncL0Args a0{};
  a0.px = px; a0.rx = rx; a0.pemb = pemb; a0.len = len;
  const int wihIdx0[4] = {8, 12, 24, 28}, bihIdx0[4] = {10, 14, 26, 30},
            bhhIdx0[4] = {11, 15, 27, 31}, whhMap0[4] = {0, 1, 4, 5};
  for (int s = 0; s < 4; s++) {
    a0.wih[s] = F(wihIdx0[s]); a0.whh[s] = whh16[whhMap0[s]];
    a0.bih[s] = F(bihIdx0[s]); a0.bhh[s] = F(bhhIdx0[s]);
    int dir = s & 1;
    a0.dir[s] = dir;
    a0.kx[s] = (s < 2) ? PV : 3;
    a0.masked[s] = (s < 2) ? 1 : 0;
    a0.isPitch[s] = (s < 2) ? 1 : 0;
    a0.out[s] = ((s < 2) ? encP : encR) + dir * 512;
    a0.hT[s] = ((s < 2) ? nPx : nRx) + dir * 512;     // slots [hf0|hb0|hf1|hb1]
  }
  enc_l0_kernel<<<32, 512, 0, stream>>>(a0);

  EncL1Args a1{};
  const int bihIdx1[4] = {18, 22, 34, 38}, bhhIdx1[4] = {19, 23, 35, 39},
            whhMap1[4] = {2, 3, 6, 7};
  a1.len = len;
  for (int s = 0; s < 4; s++) {
    a1.xin[s] = (s < 2) ? encP : encR;
    a1.wih[s] = wihL1[s]; a1.whh[s] = whh16[whhMap1[s]];
    a1.bih[s] = F(bihIdx1[s]); a1.bhh[s] = F(bhhIdx1[s]);
    int dir = s & 1;
    a1.dir[s] = dir;
    a1.masked[s] = (s < 2) ? 1 : 0;
    a1.hT[s] = ((s < 2) ? nPx : nRx) + (2 + dir) * 512;
  }
  enc_l1_kernel<<<32, 512, 0, stream>>>(a1);

  // VAE heads -> directly into d_out tail
  float* outF = (float*)d_out;
  const size_t RECON = (size_t)128 * 256 * VD;
  float* pMuO = outF + RECON;
  float* pScO = pMuO + 128 * 128;
  float* rMuO = pScO + 128 * 128;
  float* rScO = rMuO + 128 * 128;
  auto lin = [&](const float* Ap, const _Float16* Wp, const float* bp, float* Cp,
                 int M, int N, int K, int act) {
    int tiles = (M / 16) * (N / 16);
    lin_kernel<<<(tiles + 7) / 8, 256, 0, stream>>>(Ap, Wp, bp, Cp, M, N, K, act);
  };
  lin(nPx, head16[0], F(41), pMuO, 128, 128, 2048, 0);
  lin(nPx, head16[1], F(43), pScO, 128, 128, 2048, 2);   // exp
  lin(nRx, head16[2], F(45), rMuO, 128, 128, 2048, 0);
  lin(nRx, head16[3], F(47), rScO, 128, 128, 2048, 2);

  sample_kernel<<<64, 256, 0, stream>>>(pMuO, pScO, rMuO, rScO, noise_p, noise_r, zBuf);

  lin(zBuf, z2b16, F(49), hbBuf, 128, 1024, 256, 1);     // selu

  beat_kernel<<<8, 512, 0, stream>>>(hbBuf, F(50),
                                     F(51), whh16[8], F(53), F(54),
                                     beatWih1, whh16[9], F(57), F(58), beatOut);

  lin(beatOut, b2th16, F(60), hTick, 2048, 1024, 512, 1);  // selu
  lin(beatOut, b2ti16, F(62), cTick, 2048, 512, 512, 1);   // selu

  tick_kernel<<<8, 512, 0, stream>>>(hTick, cTick, F(63), demb,
                                     F(64), tickWihCt, whh16[10], F(66), F(67),
                                     tickWih1, whh16[11], F(70), F(71),
                                     t2n16, F(73), outF);
}
